// MyConv2D_69612829934289
// MI455X (gfx1250) — matrix operand order repaired and checked by reading, not compiled, not run on hardware
//
#include <hip/hip_runtime.h>

// ---------------------------------------------------------------------------
// 3x3 SAME conv, stride 1, NHWC fp32 -> im2col GEMM on CDNA5 WMMA (f16 math,
// fp32 accumulate).  M = 32*112*112 = 401408, N = 256, K = 9*128 = 1152.
// f16 chosen over bf16: inputs ~N(0,1), weights ~+-0.07 -> no range risk,
// and f16's 10-bit mantissa gives ~8x lower quantization error at identical
// V_WMMA_16X16X32 throughput.
// ---------------------------------------------------------------------------

typedef __attribute__((ext_vector_type(16))) _Float16 f16x16;
typedef __attribute__((ext_vector_type(8)))  float    f32x8;

#define IN_CH   128
#define IMG_H   112
#define IMG_W   112
#define NBATCH  32
#define FILT    256
#define KTOT    1152        // 9 * 128
#define BM      128         // output pixels per block
#define KC      64          // K-chunk (one (kh,kw), 64 contiguous channels)
#define NCHUNK  18          // 1152 / 64
#define APITCH  72          // f16 elems per LDS A row (64 + 8 pad -> 144 B)
#define BPITCH  72

__device__ __forceinline__ unsigned short f2h(float x) {
    _Float16 h = (_Float16)x;                       // v_cvt_f16_f32 (RNE)
    return __builtin_bit_cast(unsigned short, h);
}

__device__ __forceinline__ unsigned int pack_f16(float a, float b) {
    return (unsigned int)f2h(a) | ((unsigned int)f2h(b) << 16);
}

// Pre-pass: fp32 HWIO kernel (1152 x 256) -> f16 in ws, layout
// [kchunk 0..17][filter 0..255][kk 0..63]  (coalesced 128 B rows for main kernel)
__global__ void conv_prep_wgt(const float* __restrict__ krn,
                              unsigned short* __restrict__ wsB) {
    int tid = blockIdx.x * 256 + threadIdx.x;   // 0 .. 294911
    int f = tid / KTOT;
    int k = tid - f * KTOT;
    float v = krn[(size_t)k * FILT + f];
    wsB[(((unsigned)(k >> 6) * FILT + f) << 6) + (k & 63)] = f2h(v);
}

__global__ __launch_bounds__(256)
void conv2d_wmma(const float* __restrict__ in,
                 const unsigned short* __restrict__ wgt,
                 const float* __restrict__ bias,
                 float* __restrict__ out) {
    __shared__ __align__(16) unsigned short lA[BM * APITCH];    // 18 KB
    __shared__ __align__(16) unsigned short lB[FILT * BPITCH];  // 36 KB

    const int tid  = threadIdx.x;
    const int lane = tid & 31;
    const int wid  = tid >> 5;
    const int wm   = wid & 1;     // M offset = wm*64
    const int wn   = wid >> 1;    // N offset = wn*64
    const int blockM = blockIdx.x;

    // A-tile loader role: 2 threads per output-pixel row, 32 channels each
    const int arow  = tid >> 1;   // 0..127
    const int ahalf = tid & 1;    // channel half within the 64-wide chunk
    const int p  = blockM * BM + arow;
    const int n  = p / (IMG_H * IMG_W);
    const int hw = p - n * (IMG_H * IMG_W);
    const int h  = hw / IMG_W;
    const int w  = hw - h * IMG_W;

    f32x8 acc[4][4] = {};

    for (int kc = 0; kc < NCHUNK; ++kc) {
        const int kh    = kc / 6;
        const int kw    = (kc >> 1) % 3;
        const int cbase = (kc & 1) << 6;          // 0 or 64
        const int ih = h + kh - 1, iw = w + kw - 1;
        const bool valid = (ih >= 0) & (ih < IMG_H) & (iw >= 0) & (iw < IMG_W);

        // ---- stage A (fp32 -> f16) into LDS, zero-fill SAME-padding rows
        const float4* src4 = (const float4*)
            (in + (((long)n * IMG_H + ih) * IMG_W + iw) * (long)IN_CH
                + cbase + ahalf * 32);
        unsigned short* adst = &lA[arow * APITCH + ahalf * 32];
        #pragma unroll
        for (int jj = 0; jj < 4; ++jj) {
            float4 f0 = valid ? src4[2 * jj]     : make_float4(0.f, 0.f, 0.f, 0.f);
            float4 f1 = valid ? src4[2 * jj + 1] : make_float4(0.f, 0.f, 0.f, 0.f);
            uint4 q;
            q.x = pack_f16(f0.x, f0.y);  q.y = pack_f16(f0.z, f0.w);
            q.z = pack_f16(f1.x, f1.y);  q.w = pack_f16(f1.z, f1.w);
            *(uint4*)(adst + jj * 8) = q;
        }

        // ---- stage B (pre-converted f16) into LDS: thread = one filter row
        {
            const uint4* bs = (const uint4*)(wgt + ((size_t)(kc * FILT + tid) << 6));
            unsigned short* bdst = &lB[tid * BPITCH];
            #pragma unroll
            for (int j = 0; j < 8; ++j)
                *(uint4*)(bdst + j * 8) = bs[j];
        }

        // ---- prefetch next chunk's input rows (global_prefetch_b8)
        if (kc + 1 < NCHUNK) {
            const int kc2 = kc + 1;
            const int ih2 = h + kc2 / 6 - 1, iw2 = w + ((kc2 >> 1) % 3) - 1;
            if ((ih2 >= 0) & (ih2 < IMG_H) & (iw2 >= 0) & (iw2 < IMG_W)) {
                const float* pf = in + (((long)n * IMG_H + ih2) * IMG_W + iw2) * (long)IN_CH
                                     + (((kc2 & 1) << 6) + ahalf * 32);
                __builtin_prefetch(pf, 0, 1);
            }
        }

        __syncthreads();

        // ---- two K=32 WMMA steps over the staged 64-wide chunk
        #pragma unroll
        for (int ks = 0; ks < 2; ++ks) {
            const int ko = ks * 32;
            union { f16x16 v; uint4 q[2]; } a[4], b[4];
            #pragma unroll
            for (int i = 0; i < 4; ++i) {
                // ISA 16-bit A layout: lanes 0-15 -> K0-7|K16-23, lanes 16-31 -> K8-15|K24-31
                const unsigned short* pa =
                    &lA[(wm * 64 + i * 16 + (lane & 15)) * APITCH + ko + ((lane >> 4) & 1) * 8];
                a[i].q[0] = *(const uint4*)pa;
                a[i].q[1] = *(const uint4*)(pa + 16);
            }
            #pragma unroll
            for (int j = 0; j < 4; ++j) {
                // B layout: lane = column N, lanes 0-15 -> K0-7|K16-23, lanes 16-31 -> K8-15|K24-31
                const unsigned short* pb =
                    &lB[(wn * 64 + j * 16 + (lane & 15)) * BPITCH + ko + ((lane >> 4) & 1) * 8];
                b[j].q[0] = *(const uint4*)pb;
                b[j].q[1] = *(const uint4*)(pb + 16);
            }
            #pragma unroll
            for (int i = 0; i < 4; ++i)
                #pragma unroll
                for (int j = 0; j < 4; ++j)
                    acc[i][j] = __builtin_amdgcn_wmma_f32_16x16x32_f16(
                        false, a[i].v, false, b[j].v,
                        (short)0, acc[i][j], false, false);
        }
        __syncthreads();
    }

    // ---- epilogue: bias add + store per C-matrix VGPR layout
    float bj[4];
    #pragma unroll
    for (int j = 0; j < 4; ++j) bj[j] = bias[wn * 64 + j * 16 + (lane & 15)];
    const int mtop = (lane >> 4) & 1;   // lanes 16-31 hold rows M = r + 8
    #pragma unroll
    for (int i = 0; i < 4; ++i) {
        #pragma unroll
        for (int r = 0; r < 8; ++r) {
            const int M = i * 16 + r + mtop * 8;
            const size_t prow = ((size_t)blockM * BM + wm * 64 + M) * FILT;
            #pragma unroll
            for (int j = 0; j < 4; ++j)
                out[prow + wn * 64 + j * 16 + (lane & 15)] = acc[i][j][r] + bj[j];
        }
    }
}

extern "C" void kernel_launch(void* const* d_in, const int* in_sizes, int n_in,
                              void* d_out, int out_size, void* d_ws, size_t ws_size,
                              hipStream_t stream) {
    (void)in_sizes; (void)n_in; (void)out_size; (void)ws_size;
    const float* in   = (const float*)d_in[0];
    const float* krn  = (const float*)d_in[1];
    const float* bias = (const float*)d_in[2];
    float* out = (float*)d_out;
    unsigned short* wsB = (unsigned short*)d_ws;   // 1152*256*2 = 576 KB

    conv_prep_wgt<<<KTOT, 256, 0, stream>>>(krn, wsB);            // 294912 elems
    const int mtiles = (NBATCH * IMG_H * IMG_W) / BM;             // 3136
    conv2d_wmma<<<mtiles, 256, 0, stream>>>(in, wsB, bias, out);
}
